// SA_MoE_55688545960207
// MI455X (gfx1250) — matrix operand order repaired and checked by reading, not compiled, not run on hardware
//
#include <hip/hip_runtime.h>
#include <hip/hip_bf16.h>

typedef __attribute__((ext_vector_type(16))) _Float16 v16h;
typedef __attribute__((ext_vector_type(8)))  _Float16 v8h;
typedef __attribute__((ext_vector_type(8)))  float    v8f;

#define NB 8
#define NC 16
#define NS 256
#define ND 512
#define NE 8
#define NDFF 1024
#define NTOK 32768   // NB*NC*NS
#define NHID 128
#define NCOND 64

// ---------------------------------------------------------------------------
// Transpose + convert f32 [batch][rows][cols] -> f16 [batch][cols][rows]
// ---------------------------------------------------------------------------
__global__ void k_transpose_f16(const float* __restrict__ in, _Float16* __restrict__ out,
                                int rows, int cols, int batch) {
  size_t i = (size_t)blockIdx.x * blockDim.x + threadIdx.x;
  size_t per = (size_t)rows * cols;
  size_t total = per * batch;
  if (i >= total) return;
  size_t b = i / per, rem = i % per;
  int r = (int)(rem / cols), c = (int)(rem % cols);
  out[b * per + (size_t)c * rows + r] = (_Float16)in[i];
}

// ---------------------------------------------------------------------------
// Per-subject backbone hypernet (tiny): se -> h1 -> gsc -> s1
// ---------------------------------------------------------------------------
__global__ void k_hypernet_small(const int* __restrict__ subject_ids,
                                 const float* __restrict__ subj_emb,
                                 const float* __restrict__ bb_w1, const float* __restrict__ bb_b1,
                                 const float* __restrict__ bb_w2, const float* __restrict__ bb_b2,
                                 const float* __restrict__ st_w1, const float* __restrict__ st_b1,
                                 float* __restrict__ gsc_out, float* __restrict__ s1_out) {
  __shared__ float se[NB][64], h1[NB][NHID], gsc[NB][NHID];
  int tid = threadIdx.x;
  for (int i = tid; i < NB * 64; i += 256) {
    int b = i >> 6, j = i & 63;
    int id = subject_ids[b] - 1;
    id = id < 0 ? 0 : (id > 49 ? 49 : id);
    se[b][j] = subj_emb[id * 64 + j];
  }
  __syncthreads();
  for (int i = tid; i < NB * NHID; i += 256) {
    int b = i >> 7, j = i & 127;
    float s = bb_b1[j];
    for (int k = 0; k < 64; ++k) s += se[b][k] * bb_w1[k * NHID + j];
    h1[b][j] = fmaxf(s, 0.f);
  }
  __syncthreads();
  for (int i = tid; i < NB * NHID; i += 256) {
    int b = i >> 7, j = i & 127;
    float s = bb_b2[j];
    for (int k = 0; k < NHID; ++k) s += h1[b][k] * bb_w2[k * NHID + j];
    gsc[b][j] = s;
    gsc_out[i] = s;
  }
  __syncthreads();
  for (int i = tid; i < NB * 64; i += 256) {
    int b = i >> 6, j = i & 63;
    float s = st_b1[j];
    for (int k = 0; k < NHID; ++k) s += gsc[b][k] * st_w1[k * 64 + j];
    s1_out[i] = fmaxf(s, 0.f);
  }
}

// ---------------------------------------------------------------------------
// FiLM params: sp = s1 @ st_w2 + st_b2 ; gamma = softplus(sp0)+1e-8, beta = sp1
// ---------------------------------------------------------------------------
__global__ void k_film_params(const float* __restrict__ s1, const float* __restrict__ st_w2,
                              const float* __restrict__ st_b2,
                              float* __restrict__ gamma, float* __restrict__ beta) {
  int i = blockIdx.x * blockDim.x + threadIdx.x;   // over NB * C*D = 8*8192
  if (i >= NB * NC * ND) return;
  int b = i >> 13;
  int cd = i & 8191;
  int j0 = cd * 2, j1 = j0 + 1;
  const float* s = s1 + b * 64;
  float a0 = st_b2[j0], a1 = st_b2[j1];
  for (int k = 0; k < 64; ++k) {
    float sv = s[k];
    a0 += sv * st_w2[k * (NC * ND * 2) + j0];
    a1 += sv * st_w2[k * (NC * ND * 2) + j1];
  }
  float g = (a0 > 20.f) ? a0 : log1pf(expf(a0));
  gamma[i] = g + 1e-8f;
  beta[i] = a1;
}

// ---------------------------------------------------------------------------
// LayerNorm + FiLM, one wave32 per token, output f16 tokens
// ---------------------------------------------------------------------------
__global__ void k_ln_film(const float* __restrict__ x, const float* __restrict__ ln_g,
                          const float* __restrict__ ln_b, const float* __restrict__ gamma,
                          const float* __restrict__ beta, _Float16* __restrict__ t16) {
  int wave = threadIdx.x >> 5, lane = threadIdx.x & 31;
  int n = blockIdx.x * 8 + wave;
  const float* row = x + (size_t)n * ND;
  float vals[16];
  float s = 0.f, ss = 0.f;
  int base = lane * 16;
#pragma unroll
  for (int i = 0; i < 16; ++i) { float v = row[base + i]; vals[i] = v; s += v; ss += v * v; }
#pragma unroll
  for (int m = 16; m >= 1; m >>= 1) {
    s  += __shfl_xor(s, m, 32);
    ss += __shfl_xor(ss, m, 32);
  }
  float mu = s * (1.f / ND);
  float var = ss * (1.f / ND) - mu * mu;
  float rstd = rsqrtf(var + 1e-5f);
  int bc = n >> 8;                       // b*16+c
  const float* gam = gamma + (size_t)bc * ND;
  const float* bet = beta + (size_t)bc * ND;
  _Float16* out = t16 + (size_t)n * ND;
#pragma unroll
  for (int i = 0; i < 16; ++i) {
    int d = base + i;
    float v = (vals[i] - mu) * rstd * ln_g[d] + ln_b[d];
    v = gam[d] * v + bet[d];
    out[d] = (_Float16)v;
  }
}

// ---------------------------------------------------------------------------
// Router: one wave per token. softmax, top-2 gates, per-subject prob sums,
// and append token to its two experts' gather lists.
// ---------------------------------------------------------------------------
__global__ void k_router(const _Float16* __restrict__ t16, const float* __restrict__ rw,
                         const float* __restrict__ rb, float* __restrict__ gates,
                         float* __restrict__ probs_sum, int* __restrict__ counts,
                         int* __restrict__ lists) {
  int wave = threadIdx.x >> 5, lane = threadIdx.x & 31;
  int n = blockIdx.x * 8 + wave;
  const _Float16* row = t16 + (size_t)n * ND;
  float acc[NE];
#pragma unroll
  for (int e = 0; e < NE; ++e) acc[e] = 0.f;
  int k0 = lane * 16;
  for (int i = 0; i < 16; ++i) {
    float tv = (float)row[k0 + i];
    const float* w = rw + (size_t)(k0 + i) * NE;
#pragma unroll
    for (int e = 0; e < NE; ++e) acc[e] += tv * w[e];
  }
#pragma unroll
  for (int m = 16; m >= 1; m >>= 1)
#pragma unroll
    for (int e = 0; e < NE; ++e) acc[e] += __shfl_xor(acc[e], m, 32);
  float mx = -1e30f;
#pragma unroll
  for (int e = 0; e < NE; ++e) { acc[e] += rb[e]; mx = fmaxf(mx, acc[e]); }
  float p[NE]; float se = 0.f;
#pragma unroll
  for (int e = 0; e < NE; ++e) { p[e] = expf(acc[e] - mx); se += p[e]; }
  float inv = 1.f / se;
#pragma unroll
  for (int e = 0; e < NE; ++e) p[e] *= inv;
  int i0 = 0;
#pragma unroll
  for (int e = 1; e < NE; ++e) if (p[e] > p[i0]) i0 = e;
  int i1 = (i0 == 0) ? 1 : 0;
#pragma unroll
  for (int e = 0; e < NE; ++e) if (e != i0 && p[e] > p[i1]) i1 = e;
  float invs = 1.f / (p[i0] + p[i1] + 1e-8f);
  if (lane < NE)
    gates[(size_t)n * NE + lane] = (lane == i0) ? p[i0] * invs : ((lane == i1) ? p[i1] * invs : 0.f);
  if (lane == 0) {
    int b = n >> 12;
#pragma unroll
    for (int e = 0; e < NE; ++e) atomicAdd(&probs_sum[b * NE + e], p[e]);
    int pos0 = atomicAdd(&counts[i0], 1);
    lists[(size_t)i0 * NTOK + pos0] = n;
    int pos1 = atomicAdd(&counts[i1], 1);
    lists[(size_t)i1 * NTOK + pos1] = n;
  }
}

// ---------------------------------------------------------------------------
// cond / selection embedding / hyper bias (tiny, one block)
// ---------------------------------------------------------------------------
__global__ void k_cond(const float* __restrict__ probs_sum, const float* __restrict__ exp_emb,
                       const float* __restrict__ gsc, const float* __restrict__ moe_w1,
                       const float* __restrict__ moe_b1, const float* __restrict__ moe_w2,
                       const float* __restrict__ moe_b2, const float* __restrict__ hw1,
                       const float* __restrict__ hb1, float* __restrict__ hyper_bias) {
  __shared__ float un[NB][NE], cat[NB][160], m1[NB][NHID], cond[NB][NCOND];
  int tid = threadIdx.x;
  if (tid < NB) {
    int b = tid;
    float p[NE];
    for (int e = 0; e < NE; ++e) p[e] = probs_sum[b * NE + e] * (1.f / (NC * NS));
    int i0 = 0;
    for (int e = 1; e < NE; ++e) if (p[e] > p[i0]) i0 = e;
    int i1 = (i0 == 0) ? 1 : 0;
    for (int e = 0; e < NE; ++e) if (e != i0 && p[e] > p[i1]) i1 = e;
    p[i0] = 0.f; p[i1] = 0.f;
    float s = 0.f;
    for (int e = 0; e < NE; ++e) s += p[e];
    float inv = 1.f / (s + 1e-8f);
    for (int e = 0; e < NE; ++e) un[b][e] = p[e] * inv;
  }
  __syncthreads();
  for (int i = tid; i < NB * NHID; i += 256) { int b = i >> 7, j = i & 127; cat[b][j] = gsc[b * NHID + j]; }
  for (int i = tid; i < NB * 32; i += 256) {
    int b = i >> 5, j = i & 31;
    float s = 0.f;
    for (int e = 0; e < NE; ++e) s += un[b][e] * exp_emb[e * 32 + j];
    cat[b][NHID + j] = s;
  }
  __syncthreads();
  for (int i = tid; i < NB * NHID; i += 256) {
    int b = i >> 7, j = i & 127;
    float s = moe_b1[j];
    for (int k = 0; k < 160; ++k) s += cat[b][k] * moe_w1[k * NHID + j];
    m1[b][j] = fmaxf(s, 0.f);
  }
  __syncthreads();
  for (int i = tid; i < NB * NCOND; i += 256) {
    int b = i >> 6, j = i & 63;
    float s = moe_b2[j];
    for (int k = 0; k < NHID; ++k) s += m1[b][k] * moe_w2[k * NCOND + j];
    cond[b][j] = s;
  }
  __syncthreads();
  for (int i = tid; i < NB * NCOND; i += 256) {
    int b = i >> 6, j = i & 63;
    float s = hb1[j];
    for (int k = 0; k < NCOND; ++k) s += cond[b][k] * hw1[(ND + k) * 64 + j];
    hyper_bias[i] = s;
  }
}

// ---------------------------------------------------------------------------
// WMMA GEMM + bias + ReLU -> f16 (compact rows).  A [*,K] f16, BT [NF,K] f16.
// One wave computes 16 rows x 64 cols (4 N-tiles), 8 waves/block.
// list/count != null: gathered mode — A row = list[pos] (clamped for pads),
// stores masked to pos < count. bias_row_shift >= 0: per-subject row bias.
// ---------------------------------------------------------------------------
__global__ void k_wmma_gemm_relu(const _Float16* __restrict__ A, const _Float16* __restrict__ BT,
                                 const float* __restrict__ bias, _Float16* __restrict__ H,
                                 int K, int NF, int bias_row_shift,
                                 const int* __restrict__ list, const int* __restrict__ count) {
  int wave = threadIdx.x >> 5, lane = threadIdx.x & 31;
  int lm = lane & 15, lh = lane >> 4;
  int m0 = (blockIdx.y * 8 + wave) * 16;
  int n0 = blockIdx.x * 64;
  int cnt = 0x7fffffff;
  if (count) {
    cnt = *count;
    int padded = (cnt + 127) & ~127;
    if ((int)blockIdx.y * 128 >= padded) return;
  }
  int pA = m0 + lm;
  int tok = pA;
  if (list) {
    int pc = pA < cnt ? pA : (cnt - 1);
    tok = list[pc];
  }
  const _Float16* arow = A + (size_t)tok * K;
  const _Float16* brow = BT + (size_t)(n0 + lm) * K + lh * 16;
  v8f acc[4] = {};
  for (int k0 = 0; k0 < K; k0 += 32) {
    v8h alo = *(const v8h*)(arow + k0 + lh * 16);
    v8h ahi = *(const v8h*)(arow + k0 + lh * 16 + 8);
    v16h a;
#pragma unroll
    for (int i = 0; i < 8; ++i) { a[i] = alo[i]; a[8 + i] = ahi[i]; }
#pragma unroll
    for (int t = 0; t < 4; ++t) {
      v16h b = *(const v16h*)(brow + (size_t)t * 16 * K + k0);
      acc[t] = __builtin_amdgcn_wmma_f32_16x16x32_f16(false, a, false, b, (short)0, acc[t],
                                                      false, false);
    }
  }
#pragma unroll
  for (int t = 0; t < 4; ++t) {
    int col = n0 + t * 16 + lm;
#pragma unroll
    for (int r = 0; r < 8; ++r) {
      int row = m0 + r + lh * 8;
      if (row >= cnt) continue;
      int brow_idx = (bias_row_shift >= 0) ? (row >> bias_row_shift) : 0;
      float bv = (bias_row_shift >= 0) ? bias[(size_t)brow_idx * NF + col] : bias[col];
      float v = acc[t][r] + bv;
      H[(size_t)row * NF + col] = (_Float16)fmaxf(v, 0.f);
    }
  }
}

// ---------------------------------------------------------------------------
// WMMA GEMM, epilogue: out[tok,col] += gate * (acc + bias[col]).
// list/count != null: gathered mode — A rows are compact positions,
// tok = list[row], stores masked to row < count. gates==null -> gate=1.
// ---------------------------------------------------------------------------
__global__ void k_wmma_gemm_accum(const _Float16* __restrict__ A, const _Float16* __restrict__ BT,
                                  const float* __restrict__ bias, const float* __restrict__ gates,
                                  int e, float* __restrict__ out, int K, int NF,
                                  const int* __restrict__ list, const int* __restrict__ count) {
  int wave = threadIdx.x >> 5, lane = threadIdx.x & 31;
  int lm = lane & 15, lh = lane >> 4;
  int m0 = (blockIdx.y * 8 + wave) * 16;
  int n0 = blockIdx.x * 64;
  int cnt = 0x7fffffff;
  if (count) {
    cnt = *count;
    int padded = (cnt + 127) & ~127;
    if ((int)blockIdx.y * 128 >= padded) return;
  }
  const _Float16* arow = A + (size_t)(m0 + lm) * K;
  const _Float16* brow = BT + (size_t)(n0 + lm) * K + lh * 16;
  v8f acc[4] = {};
  for (int k0 = 0; k0 < K; k0 += 32) {
    v8h alo = *(const v8h*)(arow + k0 + lh * 16);
    v8h ahi = *(const v8h*)(arow + k0 + lh * 16 + 8);
    v16h a;
#pragma unroll
    for (int i = 0; i < 8; ++i) { a[i] = alo[i]; a[8 + i] = ahi[i]; }
#pragma unroll
    for (int t = 0; t < 4; ++t) {
      v16h b = *(const v16h*)(brow + (size_t)t * 16 * K + k0);
      acc[t] = __builtin_amdgcn_wmma_f32_16x16x32_f16(false, a, false, b, (short)0, acc[t],
                                                      false, false);
    }
  }
#pragma unroll
  for (int t = 0; t < 4; ++t) {
    int col = n0 + t * 16 + lm;
    float bv = bias[col];
#pragma unroll
    for (int r = 0; r < 8; ++r) {
      int row = m0 + r + lh * 8;
      if (row >= cnt) continue;
      int tok = list ? list[row] : row;
      float g = gates ? gates[(size_t)tok * NE + e] : 1.0f;
      out[(size_t)tok * NF + col] += g * (acc[t][r] + bv);
    }
  }
}

// ---------------------------------------------------------------------------
extern "C" void kernel_launch(void* const* d_in, const int* in_sizes, int n_in,
                              void* d_out, int out_size, void* d_ws, size_t ws_size,
                              hipStream_t stream) {
  const float* hidden      = (const float*)d_in[0];
  const int*   subject_ids = (const int*)d_in[1];
  const float* ln_g        = (const float*)d_in[2];
  const float* ln_b        = (const float*)d_in[3];
  const float* subj_emb    = (const float*)d_in[4];
  const float* bb_w1 = (const float*)d_in[5];  const float* bb_b1 = (const float*)d_in[6];
  const float* bb_w2 = (const float*)d_in[7];  const float* bb_b2 = (const float*)d_in[8];
  const float* st_w1 = (const float*)d_in[9];  const float* st_b1 = (const float*)d_in[10];
  const float* st_w2 = (const float*)d_in[11]; const float* st_b2 = (const float*)d_in[12];
  const float* moe_w1 = (const float*)d_in[13]; const float* moe_b1 = (const float*)d_in[14];
  const float* moe_w2 = (const float*)d_in[15]; const float* moe_b2 = (const float*)d_in[16];
  const float* exp_emb = (const float*)d_in[17];
  const float* router_w = (const float*)d_in[18]; const float* router_b = (const float*)d_in[19];
  const float* ew1 = (const float*)d_in[20]; const float* eb1 = (const float*)d_in[21];
  const float* ew2 = (const float*)d_in[22]; const float* eb2 = (const float*)d_in[23];
  const float* hw1 = (const float*)d_in[24]; const float* hb1 = (const float*)d_in[25];
  const float* hw2 = (const float*)d_in[26]; const float* hb2 = (const float*)d_in[27];
  float* out = (float*)d_out;

  char* ws = (char*)d_ws;
  size_t off = 0;
  auto alloc = [&](size_t bytes) -> void* {
    off = (off + 255) & ~(size_t)255;
    void* p = ws + off;
    off += bytes;
    return p;
  };

  _Float16* t16   = (_Float16*)alloc((size_t)NTOK * ND * 2);
  _Float16* H16   = (_Float16*)alloc((size_t)NTOK * NDFF * 2);
  _Float16* H1_16 = (_Float16*)alloc((size_t)NTOK * 64 * 2);
  _Float16* W1T   = (_Float16*)alloc((size_t)NE * ND * NDFF * 2);
  _Float16* W2T   = (_Float16*)alloc((size_t)NE * NDFF * ND * 2);
  _Float16* hw1aT = (_Float16*)alloc((size_t)64 * ND * 2);
  _Float16* hw2T  = (_Float16*)alloc((size_t)ND * 64 * 2);
  float* gamma      = (float*)alloc((size_t)NB * NC * ND * 4);
  float* beta       = (float*)alloc((size_t)NB * NC * ND * 4);
  float* gates      = (float*)alloc((size_t)NTOK * NE * 4);
  float* probs_sum  = (float*)alloc((size_t)NB * NE * 4);
  float* gsc        = (float*)alloc((size_t)NB * NHID * 4);
  float* s1         = (float*)alloc((size_t)NB * 64 * 4);
  float* hyper_bias = (float*)alloc((size_t)NB * NCOND * 4);
  int*   counts     = (int*)alloc((size_t)NE * 4);
  int*   lists      = (int*)alloc((size_t)NE * NTOK * 4);
  (void)ws_size; (void)n_in; (void)in_sizes; (void)out_size;

  // out = residual
  hipMemcpyAsync(out, hidden, (size_t)NTOK * ND * sizeof(float),
                 hipMemcpyDeviceToDevice, stream);
  hipMemsetAsync(probs_sum, 0, (size_t)NB * NE * sizeof(float), stream);
  hipMemsetAsync(counts, 0, (size_t)NE * sizeof(int), stream);

  // weight conversions (f32 -> f16 transposed [N,K])
  {
    size_t tot = (size_t)NE * ND * NDFF;
    k_transpose_f16<<<dim3((unsigned)((tot + 255) / 256)), 256, 0, stream>>>(ew1, W1T, ND, NDFF, NE);
    k_transpose_f16<<<dim3((unsigned)((tot + 255) / 256)), 256, 0, stream>>>(ew2, W2T, NDFF, ND, NE);
  }
  k_transpose_f16<<<dim3((ND * 64 + 255) / 256), 256, 0, stream>>>(hw1, hw1aT, ND, 64, 1);
  k_transpose_f16<<<dim3((64 * ND + 255) / 256), 256, 0, stream>>>(hw2, hw2T, 64, ND, 1);

  // subject hypernetwork
  k_hypernet_small<<<1, 256, 0, stream>>>(subject_ids, subj_emb, bb_w1, bb_b1, bb_w2, bb_b2,
                                          st_w1, st_b1, gsc, s1);
  k_film_params<<<(NB * NC * ND + 255) / 256, 256, 0, stream>>>(s1, st_w2, st_b2, gamma, beta);

  // tokens: LN + FiLM -> f16
  k_ln_film<<<NTOK / 8, 256, 0, stream>>>(hidden, ln_g, ln_b, gamma, beta, t16);

  // router: gates, per-subject prob sums, per-expert token gather lists
  k_router<<<NTOK / 8, 256, 0, stream>>>(t16, router_w, router_b, gates, probs_sum,
                                         counts, lists);

  // cond vector -> hyper per-subject bias
  k_cond<<<1, 256, 0, stream>>>(probs_sum, exp_emb, gsc, moe_w1, moe_b1, moe_w2, moe_b2,
                                hw1, hb1, hyper_bias);

  // MoE experts: gathered top-2 WMMA GEMMs (worst-case grid, early-exit blocks)
  for (int e = 0; e < NE; ++e) {
    const int* list_e = lists + (size_t)e * NTOK;
    k_wmma_gemm_relu<<<dim3(NDFF / 64, NTOK / 128), 256, 0, stream>>>(
        t16, W1T + (size_t)e * ND * NDFF, eb1 + (size_t)e * NDFF, H16, ND, NDFF, -1,
        list_e, counts + e);
    k_wmma_gemm_accum<<<dim3(ND / 64, NTOK / 128), 256, 0, stream>>>(
        H16, W2T + (size_t)e * NDFF * ND, eb2 + (size_t)e * ND, gates, e, out, NDFF, ND,
        list_e, counts + e);
  }

  // hyper expert: layer1 with per-subject row bias (row>>12 = subject), then layer2
  k_wmma_gemm_relu<<<dim3(1, NTOK / 128), 256, 0, stream>>>(
      t16, hw1aT, hyper_bias, H1_16, ND, 64, 12, (const int*)nullptr, (const int*)nullptr);
  k_wmma_gemm_accum<<<dim3(ND / 64, NTOK / 128), 256, 0, stream>>>(
      H1_16, hw2T, hb2, (const float*)nullptr, 0, out, 64, ND,
      (const int*)nullptr, (const int*)nullptr);
}
